// PaiDGCNN_v2_77429670412672
// MI455X (gfx1250) — compile-verified, matrix operand order checked by reading, not run on hardware
//
#include <hip/hip_runtime.h>
#include <hip/hip_bf16.h>
#include <math.h>

// ---------------------------------------------------------------------------
// Types for CDNA5 WMMA
// ---------------------------------------------------------------------------
typedef __attribute__((ext_vector_type(16))) _Float16 v16h;
typedef __attribute__((ext_vector_type(8)))  float    v8f;

#define BN_EPS 1e-5f
#define KNN_MAXK 20

#if __has_builtin(__builtin_amdgcn_s_wait_asynccnt)
#define WAIT_ASYNC(n) __builtin_amdgcn_s_wait_asynccnt(n)
#else
#define WAIT_ASYNC(n) asm volatile("s_wait_asynccnt %0" ::"i"(n) : "memory")
#endif

__device__ __forceinline__ float gelu_exact(float x) {
    return 0.5f * x * (1.0f + erff(x * 0.70710678118654752f));
}

// ---------------------------------------------------------------------------
// Fibonacci-sphere kernel table: kern[d*32 + i]
// ---------------------------------------------------------------------------
__global__ void fib_kernel(float* __restrict__ kern) {
    int i = threadIdx.x;
    if (i >= 32) return;
    double off = 2.0 / 32.0;
    double y   = (double)i * off - 1.0 + off * 0.5;
    double r   = sqrt(fmax(1.0 - y * y, 0.0));
    double inc = M_PI * (3.0 - sqrt(5.0));
    double phi = (double)((i + 1) % 32) * inc;
    kern[i]      = (float)(cos(phi) * r);
    kern[32 + i] = (float)y;
    kern[64 + i] = (float)(sin(phi) * r);
}

// ---------------------------------------------------------------------------
// x (B,3,2048) -> feat0 (B,2048,3)
// ---------------------------------------------------------------------------
__global__ void transpose_x_kernel(const float* __restrict__ x, float* __restrict__ f0) {
    int t = blockIdx.x * blockDim.x + threadIdx.x;
    int b = blockIdx.y;
    if (t >= 2048 * 3) return;
    int n = t / 3, c = t % 3;
    f0[((size_t)b * 2048 + n) * 3 + c] = x[((size_t)b * 3 + c) * 2048 + n];
}

// ---------------------------------------------------------------------------
// f32 -> f16 conversion (weights / operand precast)
// ---------------------------------------------------------------------------
__global__ void f32_to_f16_kernel(const float* __restrict__ in, _Float16* __restrict__ out,
                                  int n) {
    int i = blockIdx.x * blockDim.x + threadIdx.x;
    if (i < n) out[i] = (_Float16)in[i];
}

// ---------------------------------------------------------------------------
// KNN: queries = first Nq points, database = first Npts points of x (B,3,Nstride)
// d = -|q-p|^2, keep k largest (self first, stable ties). idx: (B,Nq,k)
// ---------------------------------------------------------------------------
__global__ void knn_kernel(const float* __restrict__ x, int Nstride, int Npts,
                           int Nq, int k, int* __restrict__ idx) {
    int i = blockIdx.x * blockDim.x + threadIdx.x;
    int b = blockIdx.y;
    if (i >= Nq) return;
    const float* xb = x + (size_t)b * 3 * Nstride;
    float qx = xb[i], qy = xb[Nstride + i], qz = xb[2 * Nstride + i];
    float bd[KNN_MAXK];
    int   bi[KNN_MAXK];
    for (int t = 0; t < k; ++t) { bd[t] = -3.0e38f; bi[t] = 0; }
    for (int j = 0; j < Npts; ++j) {
        float dx = qx - xb[j];
        float dy = qy - xb[Nstride + j];
        float dz = qz - xb[2 * Nstride + j];
        float d  = -(dx * dx + dy * dy + dz * dz);
        if (d > bd[k - 1]) {
            int p = k - 1;
            while (p > 0 && d > bd[p - 1]) {
                bd[p] = bd[p - 1]; bi[p] = bi[p - 1]; --p;
            }
            bd[p] = d; bi[p] = j;
        }
    }
    int* out = idx + ((size_t)b * Nq + i) * k;
    for (int t = 0; t < k; ++t) out[t] = bi[t];
}

// ---------------------------------------------------------------------------
// PAIConv front end: gather, geo MLP, attention weights, aggregate.
// Writes one f16 row of A (width 2*fin*32) per point.
// ---------------------------------------------------------------------------
__global__ __launch_bounds__(128)
void build_agg_kernel(const float* __restrict__ xc, int Nstride,
                      const float* __restrict__ feat, int Nsrc, int fin,
                      const int* __restrict__ idx, int K,
                      const float* __restrict__ mlp_w, const float* __restrict__ mlp_b,
                      const float* __restrict__ kern,
                      _Float16* __restrict__ Aout, int Ncur) {
    __shared__ float s_nbr[KNN_MAXK * 3];
    __shared__ float s_rel[KNN_MAXK * 3];
    __shared__ float s_dis[KNN_MAXK];
    __shared__ float s_aw[KNN_MAXK * 32];
    __shared__ float s_sum[32];
    __shared__ float s_sp[2560];   // 2*fin*K <= 2560 for all stages

    int n = blockIdx.x, b = blockIdx.y, tid = threadIdx.x;
    const int* id = idx + ((size_t)b * Ncur + n) * K;
    const float* xb = xc + (size_t)b * 3 * Nstride;

    if (tid < K) {
        int j = id[tid];
        s_nbr[tid * 3 + 0] = xb[j];
        s_nbr[tid * 3 + 1] = xb[Nstride + j];
        s_nbr[tid * 3 + 2] = xb[2 * Nstride + j];
    }
    __syncthreads();
    if (tid < K) {
        float rx = s_nbr[tid * 3 + 0] - s_nbr[0];
        float ry = s_nbr[tid * 3 + 1] - s_nbr[1];
        float rz = s_nbr[tid * 3 + 2] - s_nbr[2];
        s_rel[tid * 3 + 0] = rx;
        s_rel[tid * 3 + 1] = ry;
        s_rel[tid * 3 + 2] = rz;
        s_dis[tid] = sqrtf(rx * rx + ry * ry + rz * rz + 1e-12f);
    }
    __syncthreads();

    for (int t = tid; t < K * 32; t += blockDim.x) {
        int kk = t >> 5, m = t & 31;
        float v = s_rel[kk * 3 + 0] * kern[m]
                + s_rel[kk * 3 + 1] * kern[32 + m]
                + s_rel[kk * 3 + 2] * kern[64 + m];
        if (t == 0) v += 1.0f;
        s_aw[t] = v > 0.0f ? v : 0.0f;
    }
    __syncthreads();
    if (tid < 32) {
        float s = 0.0f;
        for (int kk = 0; kk < K; ++kk) s += s_aw[kk * 32 + tid];
        s_sum[tid] = s + 1e-6f;
    }
    __syncthreads();
    for (int t = tid; t < K * 32; t += blockDim.x) {
        float v = s_aw[t] / s_sum[t & 31];
        s_aw[t] = v * v;
    }
    __syncthreads();
    if (tid < 32) {
        float s = 0.0f;
        for (int kk = 0; kk < K; ++kk) s += s_aw[kk * 32 + tid];
        s_sum[tid] = s + 1e-6f;
    }
    __syncthreads();
    for (int t = tid; t < K * 32; t += blockDim.x) {
        float v = s_aw[t] / s_sum[t & 31];
        s_aw[t] = v > 0.1f ? v : 0.0f;
    }

    const float* fb = feat + (size_t)b * Nsrc * fin;
    for (int t = tid; t < fin * K; t += blockDim.x) {
        int c = t / K, kk = t % K;
        s_sp[c * K + kk] = fb[(size_t)id[kk] * fin + c];
    }
    for (int t = tid; t < fin * K; t += blockDim.x) {
        int c = t / K, kk = t % K;
        const float* w = mlp_w + c * 7;
        float g = w[0] * s_nbr[0] + w[1] * s_nbr[1] + w[2] * s_nbr[2]
                + w[3] * s_rel[kk * 3 + 0] + w[4] * s_rel[kk * 3 + 1]
                + w[5] * s_rel[kk * 3 + 2] + w[6] * s_dis[kk] + mlp_b[c];
        s_sp[(fin + c) * K + kk] = g;
    }
    __syncthreads();

    _Float16* arow = Aout + ((size_t)b * Ncur + n) * (size_t)(2 * fin * 32);
    for (int t = tid; t < 2 * fin * 32; t += blockDim.x) {
        int c = t >> 5, m = t & 31;
        float s = 0.0f;
        for (int kk = 0; kk < K; ++kk) s += s_sp[c * K + kk] * s_aw[kk * 32 + m];
        arow[t] = (_Float16)s;
    }
}

// ---------------------------------------------------------------------------
// WMMA GEMM, f16 operands, f32 accumulate:
//   C(M,Nc) = A(M,Kd) @ W(Nc,Kd)^T, fused bias + BN + GELU.
// Requires M%64==0, Nc%64==0, Kd%64==0.
// Block 128 threads = 4 wave32; 64x64 tile; 64-deep K tiles, double-buffered
// LDS staged with global_load_async_to_lds_b128 (ASYNCcnt).
// ---------------------------------------------------------------------------
#define LDSS 72  // halves per LDS row: 64 data + 8 pad (conflict-free frag reads)

__device__ __forceinline__ void issue_async_tile(const _Float16* __restrict__ A,
                                                 const _Float16* __restrict__ W,
                                                 int Kd, int bm, int bn, int k0,
                                                 _Float16* sA, _Float16* sB, int tid) {
    // tile = 64 rows x 64 halves (128B) => 512 x 16B chunks per matrix
#pragma unroll
    for (int it = 0; it < 4; ++it) {
        int chunk = tid + it * 128;
        int r = chunk >> 3;
        int c = (chunk & 7) * 8;  // halves
        unsigned long long ga = (unsigned long long)(A + (size_t)(bm + r) * Kd + k0 + c);
        unsigned lo = (unsigned)(uintptr_t)(sA + (size_t)r * LDSS + c);
        asm volatile("global_load_async_to_lds_b128 %0, %1, off" ::"v"(lo), "v"(ga) : "memory");
    }
#pragma unroll
    for (int it = 0; it < 4; ++it) {
        int chunk = tid + it * 128;
        int r = chunk >> 3;
        int c = (chunk & 7) * 8;
        unsigned long long ga = (unsigned long long)(W + (size_t)(bn + r) * Kd + k0 + c);
        unsigned lo = (unsigned)(uintptr_t)(sB + (size_t)r * LDSS + c);
        asm volatile("global_load_async_to_lds_b128 %0, %1, off" ::"v"(lo), "v"(ga) : "memory");
    }
}

__device__ __forceinline__ v16h load_frag(const _Float16* __restrict__ tile,
                                          int rowBase, int ks, int lane) {
    // ISA 7.12.2 16-bit A 16x32 layout: lane m = lane&15; K half-group by lane>>4
    int m = lane & 15;
    int h = (lane >> 4) * 8;
    const _Float16* p = tile + (size_t)(rowBase + m) * LDSS + ks * 32;
    v16h r;
#pragma unroll
    for (int v = 0; v < 4; ++v) {
        r[2 * v]     = p[h + 2 * v];
        r[2 * v + 1] = p[h + 2 * v + 1];
        r[8 + 2 * v]     = p[16 + h + 2 * v];
        r[8 + 2 * v + 1] = p[16 + h + 2 * v + 1];
    }
    return r;
}

__device__ __forceinline__ void store_tile(v8f acc, float* __restrict__ C, int Nc,
                                           int row0, int col0,
                                           const float* bias,
                                           const float* g, const float* bb,
                                           const float* bm, const float* bv,
                                           int do_gelu, int lane) {
    int col   = col0 + (lane & 15);
    int rbase = row0 + (lane >> 4) * 8;
    float cb = bias ? bias[col] : 0.0f;
    float sc = 1.0f, sh = 0.0f;
    if (g) {
        sc = g[col] * rsqrtf(bv[col] + BN_EPS);
        sh = bb[col] - bm[col] * sc;
    }
#pragma unroll
    for (int r = 0; r < 8; ++r) {
        float val = (acc[r] + cb) * sc + sh;
        if (do_gelu) val = gelu_exact(val);
        C[(size_t)(rbase + r) * Nc + col] = val;
    }
}

__global__ __launch_bounds__(128)
void wmma_gemm_kernel(const _Float16* __restrict__ A, const _Float16* __restrict__ W,
                      float* __restrict__ C, int M, int Nc, int Kd,
                      const float* __restrict__ bias,
                      const float* __restrict__ bn_g, const float* __restrict__ bn_b,
                      const float* __restrict__ bn_m, const float* __restrict__ bn_v,
                      int do_gelu) {
    (void)M;
    __shared__ _Float16 sA[2][64 * LDSS];
    __shared__ _Float16 sB[2][64 * LDSS];

    const int tid  = threadIdx.x;
    const int lane = tid & 31;
    const int wave = tid >> 5;
    const int bm   = blockIdx.x * 64;
    const int bn   = blockIdx.y * 64;
    const int wm   = (wave >> 1) * 32;
    const int wn   = (wave & 1) * 32;

    v8f acc00 = {}, acc01 = {}, acc10 = {}, acc11 = {};

    const int nt = Kd >> 6;  // 64-deep K tiles
    issue_async_tile(A, W, Kd, bm, bn, 0, sA[0], sB[0], tid);

    for (int t = 0; t < nt; ++t) {
        const int cur = t & 1;
        if (t + 1 < nt) {
            issue_async_tile(A, W, Kd, bm, bn, (t + 1) << 6,
                             sA[cur ^ 1], sB[cur ^ 1], tid);
            WAIT_ASYNC(8);   // only next tile's 8 ops may remain in flight
        } else {
            WAIT_ASYNC(0);
        }
        __syncthreads();     // current tile visible to all waves

#pragma unroll
        for (int ks = 0; ks < 2; ++ks) {
            v16h a0 = load_frag(sA[cur], wm + 0,  ks, lane);
            v16h a1 = load_frag(sA[cur], wm + 16, ks, lane);
            v16h b0 = load_frag(sB[cur], wn + 0,  ks, lane);
            v16h b1 = load_frag(sB[cur], wn + 16, ks, lane);
            acc00 = __builtin_amdgcn_wmma_f32_16x16x32_f16(false, a0, false, b0,
                                                           (short)0, acc00, false, false);
            acc01 = __builtin_amdgcn_wmma_f32_16x16x32_f16(false, a0, false, b1,
                                                           (short)0, acc01, false, false);
            acc10 = __builtin_amdgcn_wmma_f32_16x16x32_f16(false, a1, false, b0,
                                                           (short)0, acc10, false, false);
            acc11 = __builtin_amdgcn_wmma_f32_16x16x32_f16(false, a1, false, b1,
                                                           (short)0, acc11, false, false);
        }
        __syncthreads();     // all waves done reading before buffer is re-filled
    }

    store_tile(acc00, C, Nc, bm + wm + 0,  bn + wn + 0,  bias, bn_g, bn_b, bn_m, bn_v, do_gelu, lane);
    store_tile(acc01, C, Nc, bm + wm + 0,  bn + wn + 16, bias, bn_g, bn_b, bn_m, bn_v, do_gelu, lane);
    store_tile(acc10, C, Nc, bm + wm + 16, bn + wn + 0,  bias, bn_g, bn_b, bn_m, bn_v, do_gelu, lane);
    store_tile(acc11, C, Nc, bm + wm + 16, bn + wn + 16, bias, bn_g, bn_b, bn_m, bn_v, do_gelu, lane);
}

// ---------------------------------------------------------------------------
// Max pooling over gathered neighbors: out (B,P,f) = max_k feat[b, idx[b,p,k], :]
// ---------------------------------------------------------------------------
__global__ void pool_max_kernel(const float* __restrict__ feat, int Nsrc, int f,
                                const int* __restrict__ idx, int P, int K,
                                float* __restrict__ out) {
    int p = blockIdx.x, b = blockIdx.y;
    const int* id = idx + ((size_t)b * P + p) * K;
    const float* fb = feat + (size_t)b * Nsrc * f;
    float* ob = out + ((size_t)b * P + p) * f;
    for (int c = threadIdx.x; c < f; c += blockDim.x) {
        float m = -3.0e38f;
        for (int t = 0; t < K; ++t) {
            float v = fb[(size_t)id[t] * f + c];
            m = v > m ? v : m;
        }
        ob[c] = m;
    }
}

// ---------------------------------------------------------------------------
// Channel concat of point prefixes -> f16 h (B,64,512)
// ---------------------------------------------------------------------------
__global__ void concat_kernel(const float* __restrict__ f1p, const float* __restrict__ f2p,
                              const float* __restrict__ f3p, const float* __restrict__ f4p,
                              int P1, int P2, int P3, int P4,
                              _Float16* __restrict__ h) {
    int t = blockIdx.x * blockDim.x + threadIdx.x;
    int b = blockIdx.y;
    if (t >= 64 * 512) return;
    int p = t / 512, c = t % 512;
    float v;
    if (c < 64)       v = f1p[((size_t)b * P1 + p) * 64  + c];
    else if (c < 128) v = f2p[((size_t)b * P2 + p) * 64  + (c - 64)];
    else if (c < 256) v = f3p[((size_t)b * P3 + p) * 128 + (c - 128)];
    else              v = f4p[((size_t)b * P4 + p) * 256 + (c - 256)];
    h[((size_t)b * 64 + p) * 512 + c] = (_Float16)v;
}

// ---------------------------------------------------------------------------
// [max | mean] over P points: in (B,P,C) -> out (B, 2C)
// ---------------------------------------------------------------------------
__global__ void maxmean_kernel(const float* __restrict__ in, int P, int C,
                               float* __restrict__ out) {
    int c = blockIdx.x * blockDim.x + threadIdx.x;
    int b = blockIdx.y;
    if (c >= C) return;
    const float* ib = in + (size_t)b * P * C;
    float mx = -3.0e38f, s = 0.0f;
    for (int p = 0; p < P; ++p) {
        float v = ib[(size_t)p * C + c];
        mx = v > mx ? v : mx;
        s += v;
    }
    out[(size_t)b * 2 * C + c]     = mx;
    out[(size_t)b * 2 * C + C + c] = s / (float)P;
}

// ---------------------------------------------------------------------------
// Tiny linear for the head (M=16): out[b,o] = act(bn(in[b,:] . W[o,:] + bias))
// ---------------------------------------------------------------------------
__global__ void small_linear_kernel(const float* __restrict__ in, const float* __restrict__ W,
                                    const float* __restrict__ bias,
                                    const float* __restrict__ g, const float* __restrict__ bb,
                                    const float* __restrict__ bm, const float* __restrict__ bv,
                                    int do_gelu, float* __restrict__ out,
                                    int Kd, int Oc) {
    int o = blockIdx.x * blockDim.x + threadIdx.x;
    int b = blockIdx.y;
    if (o >= Oc) return;
    const float* ir = in + (size_t)b * Kd;
    const float* wr = W + (size_t)o * Kd;
    float s = bias ? bias[o] : 0.0f;
    for (int k = 0; k < Kd; ++k) s += ir[k] * wr[k];
    if (g) {
        float sc = g[o] * rsqrtf(bv[o] + BN_EPS);
        s = (s - bm[o]) * sc + bb[o];
    }
    if (do_gelu) s = gelu_exact(s);
    out[(size_t)b * Oc + o] = s;
}

// ---------------------------------------------------------------------------
// Orchestration
// ---------------------------------------------------------------------------
struct PaiP {
    const float *mw, *mb, *cw, *cb, *g, *b, *m, *v;
};

extern "C" void kernel_launch(void* const* d_in, const int* in_sizes, int n_in,
                              void* d_out, int out_size, void* d_ws, size_t ws_size,
                              hipStream_t stream) {
    (void)in_sizes; (void)n_in; (void)out_size; (void)ws_size;
    const int B = 16, N = 2048;

    const float* x = (const float*)d_in[0];
    PaiP P1, P2, P3, P4;
    const float** slot;
    slot = (const float**)&P1; for (int i = 0; i < 8; ++i) slot[i] = (const float*)d_in[1 + i];
    slot = (const float**)&P2; for (int i = 0; i < 8; ++i) slot[i] = (const float*)d_in[9 + i];
    slot = (const float**)&P3; for (int i = 0; i < 8; ++i) slot[i] = (const float*)d_in[17 + i];
    slot = (const float**)&P4; for (int i = 0; i < 8; ++i) slot[i] = (const float*)d_in[25 + i];
    const float* c5_w = (const float*)d_in[33];
    const float* g5 = (const float*)d_in[34];
    const float* b5 = (const float*)d_in[35];
    const float* m5 = (const float*)d_in[36];
    const float* v5 = (const float*)d_in[37];
    const float* l1_w = (const float*)d_in[38];
    const float* g6 = (const float*)d_in[39];
    const float* b6 = (const float*)d_in[40];
    const float* m6 = (const float*)d_in[41];
    const float* v6 = (const float*)d_in[42];
    const float* l2_w = (const float*)d_in[43];
    const float* l2_b = (const float*)d_in[44];
    const float* g7 = (const float*)d_in[45];
    const float* b7 = (const float*)d_in[46];
    const float* m7 = (const float*)d_in[47];
    const float* v7 = (const float*)d_in[48];
    const float* l3_w = (const float*)d_in[49];
    const float* l3_b = (const float*)d_in[50];

    // bump allocator over workspace (byte-based, 256-aligned)
    char* wp = (char*)d_ws;
    auto allocb = [&](size_t bytes) -> void* {
        void* p = wp;
        wp += (bytes + 255) & ~(size_t)255;
        return p;
    };
    float* kern  = (float*)allocb(96 * 4);
    float* feat0 = (float*)allocb((size_t)B * N * 3 * 4);
    int*   idx   = (int*)allocb((size_t)B * N * 20 * 4);
    _Float16* aggh = (_Float16*)allocb((size_t)8192 * 4096 * 2);   // max agg (pai2), f16
    _Float16* w1h  = (_Float16*)allocb((size_t)64 * 192 * 2);
    _Float16* w2h  = (_Float16*)allocb((size_t)64 * 4096 * 2);
    _Float16* w3h  = (_Float16*)allocb((size_t)128 * 4096 * 2);
    _Float16* w4h  = (_Float16*)allocb((size_t)256 * 8192 * 2);
    _Float16* w5h  = (_Float16*)allocb((size_t)1024 * 512 * 2);
    _Float16* hh   = (_Float16*)allocb((size_t)B * 64 * 512 * 2);
    float* f1    = (float*)allocb((size_t)B * 2048 * 64 * 4);
    float* f1p   = (float*)allocb((size_t)B * 512 * 64 * 4);
    float* f2    = (float*)allocb((size_t)B * 512 * 64 * 4);
    float* f2p   = (float*)allocb((size_t)B * 256 * 64 * 4);
    float* f3    = (float*)allocb((size_t)B * 256 * 128 * 4);
    float* f3p   = (float*)allocb((size_t)B * 128 * 128 * 4);
    float* f4    = (float*)allocb((size_t)B * 128 * 256 * 4);
    float* f4p   = (float*)allocb((size_t)B * 64 * 256 * 4);
    float* h5    = (float*)allocb((size_t)B * 64 * 1024 * 4);
    float* hv    = (float*)allocb((size_t)B * 2048 * 4);
    float* t1    = (float*)allocb((size_t)B * 512 * 4);
    float* t2    = (float*)allocb((size_t)B * 256 * 4);

    fib_kernel<<<1, 32, 0, stream>>>(kern);
    transpose_x_kernel<<<dim3((2048 * 3 + 255) / 256, B), 256, 0, stream>>>(x, feat0);

    // pre-convert all GEMM weights to f16 (once per launch)
    f32_to_f16_kernel<<<(64 * 192 + 255) / 256, 256, 0, stream>>>(P1.cw, w1h, 64 * 192);
    f32_to_f16_kernel<<<(64 * 4096 + 255) / 256, 256, 0, stream>>>(P2.cw, w2h, 64 * 4096);
    f32_to_f16_kernel<<<(128 * 4096 + 255) / 256, 256, 0, stream>>>(P3.cw, w3h, 128 * 4096);
    f32_to_f16_kernel<<<(256 * 8192 + 255) / 256, 256, 0, stream>>>(P4.cw, w4h, 256 * 8192);
    f32_to_f16_kernel<<<(1024 * 512 + 255) / 256, 256, 0, stream>>>(c5_w, w5h, 1024 * 512);

    // ---- Stage 1: 2048 pts, fin=3 -> 64, K=20, Kd=192
    knn_kernel<<<dim3((2048 + 63) / 64, B), 64, 0, stream>>>(x, N, 2048, 2048, 20, idx);
    build_agg_kernel<<<dim3(2048, B), 128, 0, stream>>>(x, N, feat0, 2048, 3, idx, 20,
                                                        P1.mw, P1.mb, kern, aggh, 2048);
    wmma_gemm_kernel<<<dim3(32768 / 64, 64 / 64), 128, 0, stream>>>(
        aggh, w1h, f1, 32768, 64, 192, P1.cb, P1.g, P1.b, P1.m, P1.v, 1);
    knn_kernel<<<dim3((512 + 63) / 64, B), 64, 0, stream>>>(x, N, 2048, 512, 20, idx);
    pool_max_kernel<<<dim3(512, B), 64, 0, stream>>>(f1, 2048, 64, idx, 512, 20, f1p);

    // ---- Stage 2: 512 pts, fin=64 -> 64, K=20, Kd=4096
    knn_kernel<<<dim3((512 + 63) / 64, B), 64, 0, stream>>>(x, N, 512, 512, 20, idx);
    build_agg_kernel<<<dim3(512, B), 128, 0, stream>>>(x, N, f1p, 512, 64, idx, 20,
                                                       P2.mw, P2.mb, kern, aggh, 512);
    wmma_gemm_kernel<<<dim3(8192 / 64, 64 / 64), 128, 0, stream>>>(
        aggh, w2h, f2, 8192, 64, 4096, P2.cb, P2.g, P2.b, P2.m, P2.v, 1);
    knn_kernel<<<dim3((256 + 63) / 64, B), 64, 0, stream>>>(x, N, 512, 256, 20, idx);
    pool_max_kernel<<<dim3(256, B), 64, 0, stream>>>(f2, 512, 64, idx, 256, 20, f2p);

    // ---- Stage 3: 256 pts, fin=64 -> 128, K=20, Kd=4096
    knn_kernel<<<dim3((256 + 63) / 64, B), 64, 0, stream>>>(x, N, 256, 256, 20, idx);
    build_agg_kernel<<<dim3(256, B), 128, 0, stream>>>(x, N, f2p, 256, 64, idx, 20,
                                                       P3.mw, P3.mb, kern, aggh, 256);
    wmma_gemm_kernel<<<dim3(4096 / 64, 128 / 64), 128, 0, stream>>>(
        aggh, w3h, f3, 4096, 128, 4096, P3.cb, P3.g, P3.b, P3.m, P3.v, 1);
    knn_kernel<<<dim3((128 + 63) / 64, B), 64, 0, stream>>>(x, N, 256, 128, 20, idx);
    pool_max_kernel<<<dim3(128, B), 128, 0, stream>>>(f3, 256, 128, idx, 128, 20, f3p);

    // ---- Stage 4: 128 pts, fin=128 -> 256, K=10, Kd=8192
    knn_kernel<<<dim3((128 + 63) / 64, B), 64, 0, stream>>>(x, N, 128, 128, 10, idx);
    build_agg_kernel<<<dim3(128, B), 128, 0, stream>>>(x, N, f3p, 128, 128, idx, 10,
                                                       P4.mw, P4.mb, kern, aggh, 128);
    wmma_gemm_kernel<<<dim3(2048 / 64, 256 / 64), 128, 0, stream>>>(
        aggh, w4h, f4, 2048, 256, 8192, P4.cb, P4.g, P4.b, P4.m, P4.v, 1);
    knn_kernel<<<dim3((64 + 63) / 64, B), 64, 0, stream>>>(x, N, 128, 64, 20, idx);
    pool_max_kernel<<<dim3(64, B), 256, 0, stream>>>(f4, 128, 256, idx, 64, 20, f4p);

    // ---- conv5 + global pooling + head
    concat_kernel<<<dim3((64 * 512 + 255) / 256, B), 256, 0, stream>>>(
        f1p, f2p, f3p, f4p, 512, 256, 128, 64, hh);
    wmma_gemm_kernel<<<dim3(1024 / 64, 1024 / 64), 128, 0, stream>>>(
        hh, w5h, h5, 1024, 1024, 512, nullptr, g5, b5, m5, v5, 1);
    maxmean_kernel<<<dim3((1024 + 255) / 256, B), 256, 0, stream>>>(h5, 64, 1024, hv);

    small_linear_kernel<<<dim3((512 + 63) / 64, B), 64, 0, stream>>>(
        hv, l1_w, nullptr, g6, b6, m6, v6, 1, t1, 2048, 512);
    small_linear_kernel<<<dim3((256 + 63) / 64, B), 64, 0, stream>>>(
        t1, l2_w, l2_b, g7, b7, m7, v7, 1, t2, 512, 256);
    small_linear_kernel<<<dim3(1, B), 64, 0, stream>>>(
        t2, l3_w, l3_b, nullptr, nullptr, nullptr, nullptr, 0, (float*)d_out, 256, 40);
}
